// ExpertChoiceMoE_56710748176711
// MI455X (gfx1250) — matrix operand order repaired and checked by reading, not compiled, not run on hardware
//
#include <hip/hip_runtime.h>
#include <hip/hip_bf16.h>
#include <stdint.h>

// Problem constants (B=8, T=2048 -> BT=16384)
#define BT  16384
#define DD  1024
#define NE  16
#define HH  2048
#define CAP 1280   // ceil(1.25 * 16384 / 16)

// FFN GEMM tiling: 128x64 block tile, 8 waves, each wave 32x32 (4 WMMAs/K-step)
#define MT  128
#define NT  64
#define TK  32
#define LDK 40   // padded LDS stride (elements); 40*2B = 80B keeps 16B alignment

typedef __attribute__((ext_vector_type(16))) __bf16 v16bf;
typedef __attribute__((ext_vector_type(4)))  __bf16 v4bf;
typedef __attribute__((ext_vector_type(8)))  float  v8f;

__device__ __forceinline__ float gelu_tanh(float v) {
  const float c0 = 0.7978845608028654f;   // sqrt(2/pi)
  const float c1 = 0.044715f;
  return 0.5f * v * (1.0f + tanhf(c0 * (v + c1 * v * v * v)));
}

__device__ __forceinline__ uint32_t lds_off(const void* p) {
  return (uint32_t)(uintptr_t)p;   // low 32 bits of generic LDS pointer = wave LDS byte offset
}

// CDNA5 async copy global->LDS (tracked by ASYNCcnt). VDST = LDS byte addr VGPR.
__device__ __forceinline__ void async_b128(uint32_t lds_byte_off, uint64_t gaddr) {
  asm volatile("global_load_async_to_lds_b128 %0, %1, off"
               :: "v"(lds_byte_off), "v"(gaddr) : "memory");
}
template <int N>
__device__ __forceinline__ void wait_asynccnt() {
  asm volatile("s_wait_asynccnt %0" :: "i"(N) : "memory");
}

__device__ __forceinline__ v8f wmma_bf16(v16bf a, v16bf b, v8f c) {
  return __builtin_amdgcn_wmma_f32_16x16x32_bf16(false, a, false, b, (short)0, c, false, false);
}

__global__ void moe_init_kernel(int* __restrict__ cnt) {
  if (threadIdx.x < NE) cnt[threadIdx.x] = 0;
}

// ---------------- elementwise f32 -> bf16 (for x) ----------------
__global__ __launch_bounds__(256)
void convert_x_kernel(const float* __restrict__ src, uint16_t* __restrict__ dst) {
  __bf16* d = (__bf16*)dst;
  const size_t i = ((size_t)blockIdx.x * 256 + threadIdx.x) * 4;
  const float4 v = *(const float4*)(src + i);
  v4bf o; o[0] = (__bf16)v.x; o[1] = (__bf16)v.y; o[2] = (__bf16)v.z; o[3] = (__bf16)v.w;
  *(v4bf*)(d + i) = o;
}

// ------- tiled transpose + convert: src [E][R][C] f32 -> dst [E][C][R] bf16 -------
__global__ __launch_bounds__(256)
void convert_transpose_kernel(const float* __restrict__ src, uint16_t* __restrict__ dst,
                              int R, int C) {
  __bf16* d = (__bf16*)dst;
  __shared__ float tile[64][65];
  const int e = blockIdx.z;
  const int r0 = blockIdx.x * 64, c0 = blockIdx.y * 64;
  const int tid = threadIdx.x;
#pragma unroll
  for (int t = 0; t < 16; ++t) {
    const int idx = t * 256 + tid;
    const int r = idx >> 6, c = idx & 63;
    tile[r][c] = src[((size_t)e * R + r0 + r) * C + c0 + c];
  }
  __syncthreads();
#pragma unroll
  for (int t = 0; t < 16; ++t) {
    const int idx = t * 256 + tid;
    const int c = idx >> 6, r = idx & 63;
    d[((size_t)e * C + c0 + c) * R + r0 + r] = (__bf16)tile[r][c];
  }
}

// ---------------- Gate: logits = x @ Wg, softmax, top-1 ----------------
__global__ __launch_bounds__(256)
void moe_gate_kernel(const float* __restrict__ x, const float* __restrict__ Wg,
                     float* __restrict__ tval, int* __restrict__ tidx) {
  const int t = blockIdx.x * blockDim.x + threadIdx.x;
  if (t >= BT) return;
  float acc[NE];
#pragma unroll
  for (int e = 0; e < NE; ++e) acc[e] = 0.0f;
  const float* xr = x + (size_t)t * DD;
  for (int d = 0; d < DD; ++d) {
    const float xv = xr[d];
    const float4* w4 = reinterpret_cast<const float4*>(Wg + (size_t)d * NE);
    const float4 w0 = w4[0], w1 = w4[1], w2 = w4[2], w3 = w4[3];
    acc[0]  = fmaf(xv, w0.x, acc[0]);  acc[1]  = fmaf(xv, w0.y, acc[1]);
    acc[2]  = fmaf(xv, w0.z, acc[2]);  acc[3]  = fmaf(xv, w0.w, acc[3]);
    acc[4]  = fmaf(xv, w1.x, acc[4]);  acc[5]  = fmaf(xv, w1.y, acc[5]);
    acc[6]  = fmaf(xv, w1.z, acc[6]);  acc[7]  = fmaf(xv, w1.w, acc[7]);
    acc[8]  = fmaf(xv, w2.x, acc[8]);  acc[9]  = fmaf(xv, w2.y, acc[9]);
    acc[10] = fmaf(xv, w2.z, acc[10]); acc[11] = fmaf(xv, w2.w, acc[11]);
    acc[12] = fmaf(xv, w3.x, acc[12]); acc[13] = fmaf(xv, w3.y, acc[13]);
    acc[14] = fmaf(xv, w3.z, acc[14]); acc[15] = fmaf(xv, w3.w, acc[15]);
  }
  float mx = acc[0]; int mi = 0;
#pragma unroll
  for (int e = 1; e < NE; ++e) { if (acc[e] > mx) { mx = acc[e]; mi = e; } }
  float s = 0.0f;
#pragma unroll
  for (int e = 0; e < NE; ++e) s += expf(acc[e] - mx);
  tval[t] = 1.0f / s;   // max softmax prob
  tidx[t] = mi;
}

// -------- Exact per-expert top-cap selection (top_k tie-break: lower index) --------
__global__ __launch_bounds__(256)
void moe_rank_select_kernel(const float* __restrict__ tval, const int* __restrict__ tidx,
                            int* __restrict__ cnt, int* __restrict__ list) {
  const int t = blockIdx.x * 256 + threadIdx.x;
  const float v = tval[t];
  const int   e = tidx[t];
  __shared__ float sv[256];
  __shared__ int   se[256];
  int rank = 0;
  for (int base = 0; base < BT; base += 256) {
    sv[threadIdx.x] = tval[base + threadIdx.x];
    se[threadIdx.x] = tidx[base + threadIdx.x];
    __syncthreads();
#pragma unroll 8
    for (int k = 0; k < 256; ++k) {
      const int tp = base + k;
      const bool gt = (se[k] == e) && ((sv[k] > v) || ((sv[k] == v) && (tp < t)));
      rank += gt ? 1 : 0;
    }
    __syncthreads();
  }
  if (rank < CAP) {
    const int pos = atomicAdd(&cnt[e], 1);
    list[(size_t)e * CAP + pos] = t;
  }
}

// ---------------- GEMM1: h = gelu(X_e @ W1[e] + b1[e]), bf16 out ----------------
// A: gathered token rows from xbf [BT][DD] bf16; B: W1t [E][HH][DD] bf16 (transposed)
__global__ __launch_bounds__(256)
void moe_ffn1_kernel(const uint16_t* __restrict__ xbf16, const uint16_t* __restrict__ w1t16,
                     const float* __restrict__ b1, const int* __restrict__ list,
                     const int* __restrict__ cnt, uint16_t* __restrict__ hbuf16) {
  const __bf16* xb = (const __bf16*)xbf16;
  const __bf16* wt = (const __bf16*)w1t16;
  __bf16* hbuf = (__bf16*)hbuf16;

  const int e  = blockIdx.x / (CAP / MT);
  const int m0 = (blockIdx.x % (CAP / MT)) * MT;
  const int n0 = blockIdx.y * NT;
  int me = cnt[e]; if (me > CAP) me = CAP;
  if (m0 >= me) return;                 // uniform per block

  __shared__ __bf16 As[2][MT * LDK];
  __shared__ __bf16 Bs[2][NT * LDK];

  const int tid = threadIdx.x;
  const int ra = tid >> 2;              // 0..63
  const int ka = (tid & 3) * 8;         // 0,8,16,24 (16B chunks along K)
  int r0i = m0 + ra;      if (r0i > me - 1) r0i = me - 1;
  int r1i = m0 + ra + 64; if (r1i > me - 1) r1i = me - 1;
  const int tok0 = list[(size_t)e * CAP + r0i];
  const int tok1 = list[(size_t)e * CAP + r1i];
  const uint64_t ga0 = (uint64_t)(uintptr_t)(xb + (size_t)tok0 * DD + ka);
  const uint64_t ga1 = (uint64_t)(uintptr_t)(xb + (size_t)tok1 * DD + ka);
  const uint64_t gb  = (uint64_t)(uintptr_t)(wt + ((size_t)e * HH + n0 + ra) * DD + ka);
  const uint32_t la0[2] = { lds_off(&As[0][ra * LDK + ka]),        lds_off(&As[1][ra * LDK + ka]) };
  const uint32_t la1[2] = { lds_off(&As[0][(ra + 64) * LDK + ka]), lds_off(&As[1][(ra + 64) * LDK + ka]) };
  const uint32_t lb [2] = { lds_off(&Bs[0][ra * LDK + ka]),        lds_off(&Bs[1][ra * LDK + ka]) };

  const int wave = tid >> 5, lane = tid & 31;
  const int wm = (wave & 3) * 32;       // wave's A-row base (2 x 16-row frags)
  const int wn = (wave >> 2) * 32;      // wave's col base  (2 x 16-col frags)
  const int lhalf = lane >> 4, l16 = lane & 15;

  v8f c00 = {}, c01 = {}, c10 = {}, c11 = {};

  // software pipeline: double-buffered async tiles
  async_b128(la0[0], ga0); async_b128(la1[0], ga1); async_b128(lb[0], gb);
  const int NSTAGE = DD / TK;
  for (int s = 0; s < NSTAGE; ++s) {
    const int buf = s & 1;
    const bool more = (s + 1 < NSTAGE);
    if (more) {
      const size_t kb = (size_t)(s + 1) * TK * 2;   // byte advance along K
      async_b128(la0[buf ^ 1], ga0 + kb);
      async_b128(la1[buf ^ 1], ga1 + kb);
      async_b128(lb [buf ^ 1], gb  + kb);
      wait_asynccnt<3>();               // in-order: current stage's 3 loads landed
    } else {
      wait_asynccnt<0>();
    }
    __syncthreads();
    v16bf a0, a1, b0, b1v;
    {
      const __bf16* ap0 = &As[buf][(wm + l16) * LDK + lhalf * 16];
      const __bf16* ap1 = &As[buf][(wm + 16 + l16) * LDK + lhalf * 16];
#pragma unroll
      for (int i = 0; i < 8; ++i) {
        a0[i] = ap0[i]; a0[8 + i] = ap0[8 + i];
        a1[i] = ap1[i]; a1[8 + i] = ap1[8 + i];
      }
      const __bf16* bp0 = &Bs[buf][(wn + l16) * LDK + lhalf * 16];
      const __bf16* bp1 = &Bs[buf][(wn + 16 + l16) * LDK + lhalf * 16];
#pragma unroll
      for (int i = 0; i < 16; ++i) { b0[i] = bp0[i]; b1v[i] = bp1[i]; }
    }
    c00 = wmma_bf16(a0, b0, c00);
    c01 = wmma_bf16(a0, b1v, c01);
    c10 = wmma_bf16(a1, b0, c10);
    c11 = wmma_bf16(a1, b1v, c11);
    __syncthreads();
  }

  const float bias0 = b1[(size_t)e * HH + n0 + wn + l16];
  const float bias1 = b1[(size_t)e * HH + n0 + wn + 16 + l16];
#pragma unroll
  for (int i = 0; i < 8; ++i) {
    const int gmA = m0 + wm + lhalf * 8 + i;
    const int gmB = gmA + 16;
    if (gmA < me) {
      const size_t base = ((size_t)e * CAP + gmA) * HH + n0;
      hbuf[base + wn + l16]      = (__bf16)gelu_tanh(c00[i] + bias0);
      hbuf[base + wn + 16 + l16] = (__bf16)gelu_tanh(c01[i] + bias1);
    }
    if (gmB < me) {
      const size_t base = ((size_t)e * CAP + gmB) * HH + n0;
      hbuf[base + wn + l16]      = (__bf16)gelu_tanh(c10[i] + bias0);
      hbuf[base + wn + 16 + l16] = (__bf16)gelu_tanh(c11[i] + bias1);
    }
  }
}

// ---------------- GEMM2: y[tok] = h @ W2[e] + b2[e] (scatter) ----------------
// A: hbuf [E][CAP][HH] bf16; B: W2t [E][DD][HH] bf16 (transposed)
__global__ __launch_bounds__(256)
void moe_ffn2_kernel(const uint16_t* __restrict__ hbuf16, const uint16_t* __restrict__ w2t16,
                     const float* __restrict__ b2, const int* __restrict__ list,
                     const int* __restrict__ cnt, float* __restrict__ y) {
  const __bf16* hbuf = (const __bf16*)hbuf16;
  const __bf16* wt = (const __bf16*)w2t16;

  const int e  = blockIdx.x / (CAP / MT);
  const int m0 = (blockIdx.x % (CAP / MT)) * MT;
  const int n0 = blockIdx.y * NT;
  int me = cnt[e]; if (me > CAP) me = CAP;
  if (m0 >= me) return;

  __shared__ __bf16 As[2][MT * LDK];
  __shared__ __bf16 Bs[2][NT * LDK];

  const int tid = threadIdx.x;
  const int ra = tid >> 2;
  const int ka = (tid & 3) * 8;
  int r0i = m0 + ra;      if (r0i > me - 1) r0i = me - 1;
  int r1i = m0 + ra + 64; if (r1i > me - 1) r1i = me - 1;
  const uint64_t ga0 = (uint64_t)(uintptr_t)(hbuf + ((size_t)e * CAP + r0i) * HH + ka);
  const uint64_t ga1 = (uint64_t)(uintptr_t)(hbuf + ((size_t)e * CAP + r1i) * HH + ka);
  const uint64_t gb  = (uint64_t)(uintptr_t)(wt + ((size_t)e * DD + n0 + ra) * HH + ka);
  const uint32_t la0[2] = { lds_off(&As[0][ra * LDK + ka]),        lds_off(&As[1][ra * LDK + ka]) };
  const uint32_t la1[2] = { lds_off(&As[0][(ra + 64) * LDK + ka]), lds_off(&As[1][(ra + 64) * LDK + ka]) };
  const uint32_t lb [2] = { lds_off(&Bs[0][ra * LDK + ka]),        lds_off(&Bs[1][ra * LDK + ka]) };

  const int wave = tid >> 5, lane = tid & 31;
  const int wm = (wave & 3) * 32;
  const int wn = (wave >> 2) * 32;
  const int lhalf = lane >> 4, l16 = lane & 15;

  v8f c00 = {}, c01 = {}, c10 = {}, c11 = {};

  async_b128(la0[0], ga0); async_b128(la1[0], ga1); async_b128(lb[0], gb);
  const int NSTAGE = HH / TK;
  for (int s = 0; s < NSTAGE; ++s) {
    const int buf = s & 1;
    const bool more = (s + 1 < NSTAGE);
    if (more) {
      const size_t kb = (size_t)(s + 1) * TK * 2;
      async_b128(la0[buf ^ 1], ga0 + kb);
      async_b128(la1[buf ^ 1], ga1 + kb);
      async_b128(lb [buf ^ 1], gb  + kb);
      wait_asynccnt<3>();
    } else {
      wait_asynccnt<0>();
    }
    __syncthreads();
    v16bf a0, a1, b0, b1v;
    {
      const __bf16* ap0 = &As[buf][(wm + l16) * LDK + lhalf * 16];
      const __bf16* ap1 = &As[buf][(wm + 16 + l16) * LDK + lhalf * 16];
#pragma unroll
      for (int i = 0; i < 8; ++i) {
        a0[i] = ap0[i]; a0[8 + i] = ap0[8 + i];
        a1[i] = ap1[i]; a1[8 + i] = ap1[8 + i];
      }
      const __bf16* bp0 = &Bs[buf][(wn + l16) * LDK + lhalf * 16];
      const __bf16* bp1 = &Bs[buf][(wn + 16 + l16) * LDK + lhalf * 16];
#pragma unroll
      for (int i = 0; i < 16; ++i) { b0[i] = bp0[i]; b1v[i] = bp1[i]; }
    }
    c00 = wmma_bf16(a0, b0, c00);
    c01 = wmma_bf16(a0, b1v, c01);
    c10 = wmma_bf16(a1, b0, c10);
    c11 = wmma_bf16(a1, b1v, c11);
    __syncthreads();
  }

  const float bias0 = b2[(size_t)e * DD + n0 + wn + l16];
  const float bias1 = b2[(size_t)e * DD + n0 + wn + 16 + l16];
#pragma unroll
  for (int i = 0; i < 8; ++i) {
    const int gmA = m0 + wm + lhalf * 8 + i;
    const int gmB = gmA + 16;
    if (gmA < me) {
      const int tok = list[(size_t)e * CAP + gmA];
      y[(size_t)tok * DD + n0 + wn + l16]      = c00[i] + bias0;
      y[(size_t)tok * DD + n0 + wn + 16 + l16] = c01[i] + bias1;
    }
    if (gmB < me) {
      const int tok = list[(size_t)e * CAP + gmB];
      y[(size_t)tok * DD + n0 + wn + l16]      = c10[i] + bias0;
      y[(size_t)tok * DD + n0 + wn + 16 + l16] = c11[i] + bias1;
    }
  }
}

extern "C" void kernel_launch(void* const* d_in, const int* in_sizes, int n_in,
                              void* d_out, int out_size, void* d_ws, size_t ws_size,
                              hipStream_t stream) {
  const float* x  = (const float*)d_in[0];
  const float* Wg = (const float*)d_in[1];
  const float* W1 = (const float*)d_in[2];
  const float* b1 = (const float*)d_in[3];
  const float* W2 = (const float*)d_in[4];
  const float* b2 = (const float*)d_in[5];
  float* y = (float*)d_out;

  // workspace layout (bytes)
  char* ws = (char*)d_ws;
  size_t off = 0;
  float*    tval = (float*)(ws + off);    off += (size_t)BT * 4;                 // 64 KB
  int*      tidx = (int*)(ws + off);      off += (size_t)BT * 4;                 // 64 KB
  int*      cnt  = (int*)(ws + off);      off += 256;
  int*      list = (int*)(ws + off);      off += (size_t)NE * CAP * 4;           // 80 KB
  uint16_t* xbf  = (uint16_t*)(ws + off); off += (size_t)BT * DD * 2;            // 32 MB
  uint16_t* w1t  = (uint16_t*)(ws + off); off += (size_t)NE * DD * HH * 2;       // 64 MB
  uint16_t* w2t  = (uint16_t*)(ws + off); off += (size_t)NE * DD * HH * 2;       // 64 MB
  uint16_t* hbuf = (uint16_t*)(ws + off);                                        // 80 MB

  moe_init_kernel<<<1, 64, 0, stream>>>(cnt);

  // bf16 staging: x elementwise; W1/W2 transposed so GEMM K-dim is contiguous
  convert_x_kernel<<<(BT * DD) / (256 * 4), 256, 0, stream>>>(x, xbf);
  convert_transpose_kernel<<<dim3(DD / 64, HH / 64, NE), 256, 0, stream>>>(W1, w1t, DD, HH);
  convert_transpose_kernel<<<dim3(HH / 64, DD / 64, NE), 256, 0, stream>>>(W2, w2t, HH, DD);

  moe_gate_kernel<<<BT / 256, 256, 0, stream>>>(x, Wg, tval, tidx);
  moe_rank_select_kernel<<<BT / 256, 256, 0, stream>>>(tval, tidx, cnt, list);

  // identity pass-through for unselected tokens
  hipMemcpyAsync(y, x, (size_t)BT * DD * sizeof(float), hipMemcpyDeviceToDevice, stream);

  dim3 g1(NE * (CAP / MT), HH / NT);
  moe_ffn1_kernel<<<g1, 256, 0, stream>>>(xbf, w1t, b1, list, cnt, hbuf);
  dim3 g2(NE * (CAP / MT), DD / NT);
  moe_ffn2_kernel<<<g2, 256, 0, stream>>>(hbuf, w2t, b2, list, cnt, y);
}
